// AdaptiveSoftmax_14903536517669
// MI455X (gfx1250) — compile-verified, matrix operand order checked by reading, not run on hardware
//
#include <hip/hip_runtime.h>
#include <hip/hip_bf16.h>
#include <math.h>

typedef __bf16 bf16;
typedef __attribute__((ext_vector_type(16))) __bf16 v16bf;
typedef __attribute__((ext_vector_type(8)))  float  v8f;

#define UNITS     50257
#define CUT0      2000
#define CUT1      10000
#define C_DIM     1024
#define N_TOK     4096
#define ROOT_V    2002
#define ROOT_VPAD 2048
#define T0_V      8000
#define T0_VPAD   8000
#define T0_K      256
#define T1_V      40257
#define T1_VPAD   40320
#define T1_K      64
#define VCHUNK    2048

// ---------------------------------------------------------------------------
// WMMA fragment load.  A (16x32 bf16, MxK) layout: lane&15 = M row, lane>>4
// selects K-half; VGPR0..3 hold K = h*8..h*8+7 (packed pairs), VGPR4..7 hold
// K = 16+h*8..16+h*8+7.  B (32x16, KxN) is symmetric with Bt row-major, so we
// store W transposed [Vpad][K] and use the identical loader (rowsel = N).
// Consecutive K is contiguous in memory -> two 16-byte loads per fragment.
// ---------------------------------------------------------------------------
__device__ __forceinline__ v16bf load_frag(const bf16* __restrict__ base, int ld,
                                           int rowsel, int k, int half)
{
    const bf16* p = base + (size_t)rowsel * ld + k + half * 8;
    union { v16bf v; float4 q[2]; } u;
    u.q[0] = *(const float4*)(p);        // K = k+h*8   .. +7
    u.q[1] = *(const float4*)(p + 16);   // K = k+16+h*8.. +7
    return u.v;
}

// ---------------------------------------------------------------------------
// elementwise f32 -> bf16
// ---------------------------------------------------------------------------
__global__ void conv_bf16(const float* __restrict__ in, bf16* __restrict__ out, int n)
{
    int i = blockIdx.x * 256 + threadIdx.x;
    if (i < n) out[i] = (bf16)in[i];
}

// ---------------------------------------------------------------------------
// W [K x V] f32 (row-major) -> Wt [Vpad x K] bf16, zero-padded rows.
// LDS-tiled 32x32 transpose: coalesced on both sides, 33-pitch kills bank
// conflicts.  Grid: (Vpad/32, K/32); K is always a multiple of 32 here.
// ---------------------------------------------------------------------------
__global__ __launch_bounds__(256)
void transpose_conv(const float* __restrict__ W, bf16* __restrict__ Wt,
                    int K, int V, int Vpad)
{
    __shared__ float tile[32][33];
    const int vt = blockIdx.x * 32;
    const int kt = blockIdx.y * 32;
    const int tx = threadIdx.x & 31;
    const int ty = threadIdx.x >> 5;    // 0..7

#pragma unroll
    for (int j = 0; j < 4; ++j) {
        int k = kt + ty + 8 * j;
        int v = vt + tx;
        float val = (v < V) ? W[(size_t)k * V + v] : 0.0f;   // k always < K
        tile[ty + 8 * j][tx] = val;
    }
    __syncthreads();
#pragma unroll
    for (int j = 0; j < 4; ++j) {
        int v = vt + ty + 8 * j;
        int k = kt + tx;
        Wt[(size_t)v * K + k] = (bf16)tile[tx][ty + 8 * j]; // v always < Vpad
    }
}

// ---------------------------------------------------------------------------
// C = X[N x K]bf16 @ Wt[V x K]bf16 -> outF (f32) and outB (bf16), full store.
// Block: 256 threads = 8 waves, 2 (rows) x 4 (cols) of 16x16 WMMA tiles.
// ---------------------------------------------------------------------------
__global__ __launch_bounds__(256)
void gemm_store(const bf16* __restrict__ X, const bf16* __restrict__ Wt,
                int K, int V, float* __restrict__ outF, bf16* __restrict__ outB)
{
    const int tid  = threadIdx.x;
    const int lane = tid & 31;
    const int wave = tid >> 5;
    const int wrow = wave >> 2;   // 0..1
    const int wcol = wave & 3;    // 0..3
    const int half = lane >> 4;
    const int l15  = lane & 15;

    const int m0 = blockIdx.x * 32 + wrow * 16;
    const int n0 = blockIdx.y * 64 + wcol * 16;
    const int arow = m0 + l15;
    const int brow = n0 + l15;

    v8f acc = {};
    for (int k = 0; k < K; k += 32) {
        v16bf a = load_frag(X,  K, arow, k, half);
        v16bf b = load_frag(Wt, K, brow, k, half);
        acc = __builtin_amdgcn_wmma_f32_16x16x32_bf16(false, a, false, b,
                                                      (short)0, acc, false, false);
    }
    const int ncol = n0 + l15;
#pragma unroll
    for (int r = 0; r < 8; ++r) {
        int row = m0 + half * 8 + r;           // C layout: lanes16-31 -> M=8+r
        float v = acc[r];
        outF[(size_t)row * V + ncol] = v;
        outB[(size_t)row * V + ncol] = (bf16)v;
    }
}

// ---------------------------------------------------------------------------
// Fused GEMM + online logsumexp partials, compile-time K.
// When HOIST (K<=256) the per-lane A fragments for the *entire* K extent are
// kept in registers across the whole V-chunk, so the slab loop issues only
// B-tile loads: for the FLOP-dominant tail GEMMs this halves vmem issue per
// WMMA.  Padded columns (>= Vreal) are excluded from the running (max, sum).
// ---------------------------------------------------------------------------
template<int K, bool HOIST>
__global__ __launch_bounds__(256)
void gemm_lse_t(const bf16* __restrict__ X, const bf16* __restrict__ Wt,
                int Vreal, int Vpad, int NG,
                float* __restrict__ pmax, float* __restrict__ psum)
{
    __shared__ float smax[32][64];
    __shared__ float ssum[32][64];

    const int tid  = threadIdx.x;
    const int lane = tid & 31;
    const int wave = tid >> 5;
    const int wrow = wave >> 2;
    const int wcol = wave & 3;
    const int half = lane >> 4;
    const int l15  = lane & 15;

    const int m0   = blockIdx.x * 32 + wrow * 16;
    const int v0   = blockIdx.y * VCHUNK;
    const int vhi  = min(v0 + VCHUNK, Vpad);
    const int arow = m0 + l15;

    constexpr int KSTEPS = K / 32;
    v16bf afrag[HOIST ? KSTEPS : 1];
    if constexpr (HOIST) {
#pragma unroll
        for (int ki = 0; ki < KSTEPS; ++ki)
            afrag[ki] = load_frag(X, K, arow, ki * 32, half);
    }

    float rmax[8], rsum[8];
#pragma unroll
    for (int r = 0; r < 8; ++r) { rmax[r] = -INFINITY; rsum[r] = 0.0f; }

    for (int ns = v0; ns < vhi; ns += 64) {
        const int n0   = ns + wcol * 16;
        const int brow = n0 + l15;
        v8f acc = {};
#pragma unroll
        for (int ki = 0; ki < KSTEPS; ++ki) {
            v16bf a;
            if constexpr (HOIST) a = afrag[ki];
            else                 a = load_frag(X, K, arow, ki * 32, half);
            v16bf b = load_frag(Wt, K, brow, ki * 32, half);
            acc = __builtin_amdgcn_wmma_f32_16x16x32_bf16(false, a, false, b,
                                                          (short)0, acc, false, false);
        }
        const int ncol = n0 + l15;
        if (ncol < Vreal) {
#pragma unroll
            for (int r = 0; r < 8; ++r) {
                float v = acc[r];
                if (v > rmax[r]) {
                    rsum[r] = rsum[r] * __expf(rmax[r] - v) + 1.0f;
                    rmax[r] = v;
                } else {
                    rsum[r] += __expf(v - rmax[r]);
                }
            }
        }
    }

    const int slot = wcol * 16 + l15;          // 0..63 contributions per row
#pragma unroll
    for (int r = 0; r < 8; ++r) {
        int rl = wrow * 16 + half * 8 + r;     // 0..31 local row
        smax[rl][slot] = rmax[r];
        ssum[rl][slot] = rsum[r];
    }
    __syncthreads();

    if (tid < 32) {
        float M = -INFINITY;
        for (int j = 0; j < 64; ++j) M = fmaxf(M, smax[tid][j]);
        float S = 0.0f;
        for (int j = 0; j < 64; ++j) {
            float m = smax[tid][j];
            if (m > -INFINITY) S += ssum[tid][j] * __expf(m - M);
        }
        int row = blockIdx.x * 32 + tid;
        pmax[(size_t)row * NG + blockIdx.y] = M;
        psum[(size_t)row * NG + blockIdx.y] = S;
    }
}

// combine partials across vgroups -> lse per row
__global__ void reduce_lse(const float* __restrict__ pmax, const float* __restrict__ psum,
                           int NG, float* __restrict__ lse)
{
    int row = blockIdx.x * 256 + threadIdx.x;
    if (row >= N_TOK) return;
    float M = -INFINITY;
    for (int g = 0; g < NG; ++g) M = fmaxf(M, pmax[(size_t)row * NG + g]);
    float S = 0.0f;
    for (int g = 0; g < NG; ++g)
        S += psum[(size_t)row * NG + g] * __expf(pmax[(size_t)row * NG + g] - M);
    lse[row] = M + __logf(S);
}

// ---------------------------------------------------------------------------
// picked logit: one wave per row, dot(X[row,:], W[:, col]) with W row-major
// [K x ldW].  mode 0: root target remap; mode 1: tail clip(t-lo, 0, hi-lo-1).
// ---------------------------------------------------------------------------
__global__ __launch_bounds__(256)
void picked_dot(const float* __restrict__ X, int K,
                const float* __restrict__ W, int ldW,
                const int* __restrict__ targets,
                int mode, int lo, int hi, float* __restrict__ out)
{
    int row  = blockIdx.x * 8 + (threadIdx.x >> 5);
    int lane = threadIdx.x & 31;
    int t = targets[row];
    int col;
    if (mode == 0) {
        col = (t < CUT0) ? t : ((t < CUT1) ? CUT0 : CUT0 + 1);
    } else {
        col = t - lo;
        if (col < 0) col = 0;
        int mx = hi - lo - 1;
        if (col > mx) col = mx;
    }
    const float* xr = X + (size_t)row * K;
    float s = 0.0f;
    for (int k = lane; k < K; k += 32)
        s += xr[k] * W[(size_t)k * ldW + col];
    for (int off = 16; off > 0; off >>= 1)
        s += __shfl_xor(s, off, 32);
    if (lane == 0) out[row] = s;
}

// ---------------------------------------------------------------------------
// final scalar: mean(root_ce) and masked means of the two tails, then mean/3.
// ---------------------------------------------------------------------------
__global__ __launch_bounds__(256)
void final_loss(const float* __restrict__ lseR, const float* __restrict__ pR,
                const float* __restrict__ lse0, const float* __restrict__ p0,
                const float* __restrict__ lse1, const float* __restrict__ p1,
                const int* __restrict__ targets, float* __restrict__ out)
{
    __shared__ float sh[5][256];
    float sR = 0, s0 = 0, s1 = 0, c0 = 0, c1 = 0;
    for (int i = threadIdx.x; i < N_TOK; i += 256) {
        int t = targets[i];
        sR += lseR[i] - pR[i];
        if (t >= CUT0 && t < CUT1) { s0 += lse0[i] - p0[i]; c0 += 1.0f; }
        if (t >= CUT1)             { s1 += lse1[i] - p1[i]; c1 += 1.0f; }
    }
    sh[0][threadIdx.x] = sR; sh[1][threadIdx.x] = s0; sh[2][threadIdx.x] = s1;
    sh[3][threadIdx.x] = c0; sh[4][threadIdx.x] = c1;
    __syncthreads();
    for (int st = 128; st > 0; st >>= 1) {
        if (threadIdx.x < st)
            for (int j = 0; j < 5; ++j)
                sh[j][threadIdx.x] += sh[j][threadIdx.x + st];
        __syncthreads();
    }
    if (threadIdx.x == 0) {
        float lr = sh[0][0] / (float)N_TOK;
        float l0 = sh[1][0] / fmaxf(sh[3][0], 1.0f);
        float l1 = sh[2][0] / fmaxf(sh[4][0], 1.0f);
        out[0] = (lr + l0 + l1) / 3.0f;
    }
}

// ---------------------------------------------------------------------------
extern "C" void kernel_launch(void* const* d_in, const int* in_sizes, int n_in,
                              void* d_out, int out_size, void* d_ws, size_t ws_size,
                              hipStream_t stream)
{
    const float* logits      = (const float*)d_in[0];
    const int*   targets     = (const int*)  d_in[1];
    const float* head_kernel = (const float*)d_in[2];
    const float* proj0       = (const float*)d_in[3];
    const float* scale0      = (const float*)d_in[4];
    const float* proj1       = (const float*)d_in[5];
    const float* scale1      = (const float*)d_in[6];

    char*  ws  = (char*)d_ws;
    size_t off = 0;
    auto alloc = [&](size_t bytes) -> void* {
        void* p = ws + off;
        off = (off + bytes + 255) & ~(size_t)255;
        return p;
    };

    bf16* Xb  = (bf16*)alloc((size_t)N_TOK * C_DIM * 2);
    bf16* HkT = (bf16*)alloc((size_t)ROOT_VPAD * C_DIM * 2);
    bf16* p0T = (bf16*)alloc((size_t)T0_K * C_DIM * 2);
    bf16* p1T = (bf16*)alloc((size_t)T1_K * C_DIM * 2);
    bf16* s0T = (bf16*)alloc((size_t)T0_VPAD * T0_K * 2);
    bf16* s1T = (bf16*)alloc((size_t)T1_VPAD * T1_K * 2);
    float* h0f = (float*)alloc((size_t)N_TOK * T0_K * 4);
    bf16*  h0b = (bf16*) alloc((size_t)N_TOK * T0_K * 2);
    float* h1f = (float*)alloc((size_t)N_TOK * T1_K * 4);
    bf16*  h1b = (bf16*) alloc((size_t)N_TOK * T1_K * 2);

    const int NGr = 1;
    const int NG0 = (T0_VPAD + VCHUNK - 1) / VCHUNK;   // 4
    const int NG1 = (T1_VPAD + VCHUNK - 1) / VCHUNK;   // 20
    float* pmaxR = (float*)alloc((size_t)N_TOK * NGr * 4);
    float* psumR = (float*)alloc((size_t)N_TOK * NGr * 4);
    float* pmax0 = (float*)alloc((size_t)N_TOK * NG0 * 4);
    float* psum0 = (float*)alloc((size_t)N_TOK * NG0 * 4);
    float* pmax1 = (float*)alloc((size_t)N_TOK * NG1 * 4);
    float* psum1 = (float*)alloc((size_t)N_TOK * NG1 * 4);
    float* lseR = (float*)alloc((size_t)N_TOK * 4);
    float* lse0 = (float*)alloc((size_t)N_TOK * 4);
    float* lse1 = (float*)alloc((size_t)N_TOK * 4);
    float* pkR  = (float*)alloc((size_t)N_TOK * 4);
    float* pk0  = (float*)alloc((size_t)N_TOK * 4);
    float* pk1  = (float*)alloc((size_t)N_TOK * 4);
    if (off > ws_size) return;  // insufficient workspace (deterministic no-op)

    // 1) stage bf16 activations / transposed weights
    {
        int n = N_TOK * C_DIM;
        conv_bf16<<<(n + 255) / 256, 256, 0, stream>>>(logits, Xb, n);
    }
    transpose_conv<<<dim3(ROOT_VPAD / 32, C_DIM / 32), 256, 0, stream>>>(head_kernel, HkT, C_DIM, ROOT_V, ROOT_VPAD);
    transpose_conv<<<dim3(T0_K / 32,      C_DIM / 32), 256, 0, stream>>>(proj0, p0T, C_DIM, T0_K, T0_K);
    transpose_conv<<<dim3(T1_K / 32,      C_DIM / 32), 256, 0, stream>>>(proj1, p1T, C_DIM, T1_K, T1_K);
    transpose_conv<<<dim3(T0_VPAD / 32,   T0_K / 32),  256, 0, stream>>>(scale0, s0T, T0_K, T0_V, T0_VPAD);
    transpose_conv<<<dim3(T1_VPAD / 32,   T1_K / 32),  256, 0, stream>>>(scale1, s1T, T1_K, T1_V, T1_VPAD);

    // 2) hidden projections (WMMA), materialized f32 + bf16
    gemm_store<<<dim3(N_TOK / 32, T0_K / 64), 256, 0, stream>>>(Xb, p0T, C_DIM, T0_K, h0f, h0b);
    gemm_store<<<dim3(N_TOK / 32, T1_K / 64), 256, 0, stream>>>(Xb, p1T, C_DIM, T1_K, h1f, h1b);

    // 3) fused GEMM + online logsumexp (logits never written to memory)
    gemm_lse_t<C_DIM, false><<<dim3(N_TOK / 32, NGr), 256, 0, stream>>>(Xb,  HkT, ROOT_V, ROOT_VPAD, NGr, pmaxR, psumR);
    gemm_lse_t<T0_K,  true ><<<dim3(N_TOK / 32, NG0), 256, 0, stream>>>(h0b, s0T, T0_V,   T0_VPAD,   NG0, pmax0, psum0);
    gemm_lse_t<T1_K,  true ><<<dim3(N_TOK / 32, NG1), 256, 0, stream>>>(h1b, s1T, T1_V,   T1_VPAD,   NG1, pmax1, psum1);

    reduce_lse<<<(N_TOK + 255) / 256, 256, 0, stream>>>(pmaxR, psumR, NGr, lseR);
    reduce_lse<<<(N_TOK + 255) / 256, 256, 0, stream>>>(pmax0, psum0, NG0, lse0);
    reduce_lse<<<(N_TOK + 255) / 256, 256, 0, stream>>>(pmax1, psum1, NG1, lse1);

    // 4) picked logits (f32 dot per row, one wave each)
    picked_dot<<<N_TOK / 8, 256, 0, stream>>>(logits, C_DIM, head_kernel, ROOT_V, targets, 0, 0, 0, pkR);
    picked_dot<<<N_TOK / 8, 256, 0, stream>>>(h0f, T0_K, scale0, T0_V, targets, 1, CUT0, CUT1, pk0);
    picked_dot<<<N_TOK / 8, 256, 0, stream>>>(h1f, T1_K, scale1, T1_V, targets, 1, CUT1, UNITS, pk1);

    // 5) final scalar
    final_loss<<<1, 256, 0, stream>>>(lseR, pkR, lse0, pk0, lse1, pk1, targets, (float*)d_out);
}
